// Decoder_1623497638559
// MI455X (gfx1250) — compile-verified, matrix operand order checked by reading, not run on hardware
//
#include <hip/hip_runtime.h>
#include <hip/hip_bf16.h>

typedef __attribute__((ext_vector_type(16))) __bf16 v16bf;
typedef __attribute__((ext_vector_type(8)))  __bf16 v8bf;
typedef __attribute__((ext_vector_type(8)))  float  v8f;
typedef int v4i __attribute__((vector_size(16)));

enum { F_GELU = 1, F_RES = 2, F_MASKPRE = 4, F_MASKPOST = 8, F_CONV = 16,
       F_WY = 32, F_WT = 64, F_WB16 = 128 };

__device__ __forceinline__ __bf16 tobf(float f) { return (__bf16)f; }

__device__ __forceinline__ float gelu_f(float x) {
    return 0.5f * x * (1.0f + erff(x * 0.70710678118654752f));
}

#define KMAP(e, half) ((e) + 8 * (half) + (((e) & 8) ? 8 : 0))
#define SHUF16(lo, hi) __builtin_shufflevector((lo), (hi), 0,1,2,3,4,5,6,7,8,9,10,11,12,13,14,15)

#if __has_builtin(__builtin_amdgcn_global_load_async_to_lds_b128)
#define HAVE_ASYNC_LDS 1
#endif

// Double-buffered k-loop: next iteration's A/B fragments are loaded before the
// current WMMAs consume theirs (loads get a full WMMA-step of latency cover).
// SEL: apply per-tile zero-select (conv boundary waves only).
template<bool SEL>
__device__ __forceinline__ void gemm_kloop(const __bf16* __restrict__ wr,
                                           const __bf16* const* xb,
                                           const bool* ok, int K, v8f* acc)
{
    v8bf al = *(const v8bf*)(wr);
    v8bf ah = *(const v8bf*)(wr + 16);
    v8bf bl[4], bh[4];
#pragma unroll
    for (int nt = 0; nt < 4; ++nt) {
        bl[nt] = *(const v8bf*)(xb[nt]);
        bh[nt] = *(const v8bf*)(xb[nt] + 16);
    }
    for (int k0 = 0; k0 < K; k0 += 32) {
        const int kn = (k0 + 32 < K) ? (k0 + 32) : 0;   // wrap: harmless final prefetch
        const v8bf al2 = *(const v8bf*)(wr + kn);
        const v8bf ah2 = *(const v8bf*)(wr + kn + 16);
        v8bf bl2[4], bh2[4];
#pragma unroll
        for (int nt = 0; nt < 4; ++nt) {
            bl2[nt] = *(const v8bf*)(xb[nt] + kn);
            bh2[nt] = *(const v8bf*)(xb[nt] + kn + 16);
        }
        const v16bf af = SHUF16(al, ah);
#pragma unroll
        for (int nt = 0; nt < 4; ++nt) {
            v16bf bf = SHUF16(bl[nt], bh[nt]);
            if (SEL) { const v16bf zf = {}; bf = ok[nt] ? bf : zf; }
            acc[nt] = __builtin_amdgcn_wmma_f32_16x16x32_bf16(false, af, false, bf, (short)0, acc[nt], false, false);
        }
        al = al2; ah = ah2;
#pragma unroll
        for (int nt = 0; nt < 4; ++nt) { bl[nt] = bl2[nt]; bh[nt] = bh2[nt]; }
    }
}

// Batched pointwise / dilated-conv3 GEMM. All operands bf16:
//   A: pre-packed bf16 weights (row-major; conv: (3,M,K) j-planes)
//   B: XT bf16 time-major (B,T,K)
// Wave tile 16(M) x 64(N); inner loop = 2 b128 (A) + 8 b128 (B) + 4 WMMA.
__global__ __launch_bounds__(256)
void k_gemm(const __bf16* __restrict__ Wb, const float* __restrict__ bias,
            const __bf16* __restrict__ XT,
            float* __restrict__ Y, __bf16* __restrict__ YT, __bf16* __restrict__ YB,
            const float* __restrict__ R, const float* __restrict__ mask,
            int M, int K, int T, int dil, int flags, float alpha)
{
    const int lane = threadIdx.x & 31;
    const int wid  = threadIdx.x >> 5;
    const int half = lane >> 4;
    const int col  = lane & 15;
    const int b    = blockIdx.z;
    const int m0   = blockIdx.y * 16;
    const int n0   = blockIdx.x * 512 + wid * 64;
    const int arow = m0 + col;
    const unsigned bT = (unsigned)b * T;

    v8f acc[4];
#pragma unroll
    for (int i = 0; i < 4; ++i) { v8f z = {}; acc[i] = z; }

    if (flags & F_CONV) {
        for (int j = 0; j < 3; ++j) {
            const int toff = (j - 1) * dil;
            const __bf16* wr = Wb + ((unsigned)j * M + arow) * K + 8 * half;
            const __bf16* xb[4];
            bool ok[4];
#pragma unroll
            for (int nt = 0; nt < 4; ++nt) {
                const int t = n0 + nt * 16 + col + toff;
                ok[nt] = (t >= 0) && (t < T);
                const int tc = ok[nt] ? t : 0;                 // clamped address
                xb[nt] = XT + (unsigned)(bT + tc) * K + 8 * half;
                __builtin_prefetch(xb[nt], 0, 3);              // global_prefetch_b8
            }
            // Wave-uniform interior test: scalar branch, select-free fast path.
            if (n0 + toff >= 0 && n0 + 63 + toff < T)
                gemm_kloop<false>(wr, xb, ok, K, acc);
            else
                gemm_kloop<true>(wr, xb, ok, K, acc);
        }
    } else {
        const __bf16* wr = Wb + (unsigned)arow * K + 8 * half;
        const __bf16* xb[4];
        bool ok[4] = {true, true, true, true};
#pragma unroll
        for (int nt = 0; nt < 4; ++nt) {
            xb[nt] = XT + (unsigned)(bT + n0 + nt * 16 + col) * K + 8 * half;
            __builtin_prefetch(xb[nt], 0, 3);
        }
        gemm_kloop<false>(wr, xb, ok, K, acc);
    }

#pragma unroll
    for (int nt = 0; nt < 4; ++nt) {
        const int t = n0 + nt * 16 + col;
        float mk = 1.0f;
        if (flags & (F_MASKPRE | F_MASKPOST)) mk = mask[bT + t];
        float yv[8];
#pragma unroll
        for (int v = 0; v < 8; ++v) {
            const int m = m0 + v + 8 * half;
            float y = acc[nt][v] + bias[m];
            if (flags & F_GELU) y = gelu_f(y);
            y *= alpha;
            if (flags & F_MASKPRE) y *= mk;
            if (flags & F_RES) y += R[((unsigned)b * M + m) * T + t];
            if (flags & F_MASKPOST) y *= mk;
            yv[v] = y;
        }
        if (flags & F_WY) {
#pragma unroll
            for (int v = 0; v < 8; ++v)
                Y[((unsigned)b * M + m0 + v + 8 * half) * T + t] = yv[v];
        }
        if (flags & F_WT) {
            v8bf pk;
#pragma unroll
            for (int v = 0; v < 8; ++v) pk[v] = tobf(yv[v]);
            *(v8bf*)(YT + (unsigned)(bT + t) * M + m0 + 8 * half) = pk;
        }
        if (flags & F_WB16) {
#pragma unroll
            for (int v = 0; v < 8; ++v)
                YB[((unsigned)b * M + m0 + v + 8 * half) * T + t] = tobf(yv[v]);
        }
    }
}

// LDS 32x32 tile transpose + fp32->bf16 + channel pad:
//   X (B,Cin,T) -> XT (B,T,Cpad), zero-filling channels >= Cin.
// Uses ASYNCcnt-tracked global->LDS async copies when available.
__global__ __launch_bounds__(256)
void k_pack(const float* __restrict__ X, __bf16* __restrict__ XT, int Cin, int Cpad, int T)
{
    __shared__ float tilebuf[32][36];   // 144B rows: 16B-aligned b128 targets
    const int b  = blockIdx.z;
    const int c0 = blockIdx.y * 32;
    const int t0 = blockIdx.x * 32;
    const int cl = threadIdx.x >> 3;          // 0..31 channel in tile
    const int t4 = (threadIdx.x & 7) * 4;     // 0,4,..,28 time in tile
    const int c  = c0 + cl;
    if (c < Cin) {
        const float* src = X + ((unsigned)b * Cin + c) * T + t0 + t4;
#ifdef HAVE_ASYNC_LDS
        __builtin_amdgcn_global_load_async_to_lds_b128(
            (__attribute__((address_space(1))) v4i*)src,
            (__attribute__((address_space(3))) v4i*)&tilebuf[cl][t4], 0, 0);
#else
        const float4 vsrc = *(const float4*)src;
        tilebuf[cl][t4 + 0] = vsrc.x; tilebuf[cl][t4 + 1] = vsrc.y;
        tilebuf[cl][t4 + 2] = vsrc.z; tilebuf[cl][t4 + 3] = vsrc.w;
#endif
    } else {
        tilebuf[cl][t4 + 0] = 0.0f; tilebuf[cl][t4 + 1] = 0.0f;
        tilebuf[cl][t4 + 2] = 0.0f; tilebuf[cl][t4 + 3] = 0.0f;
    }
#ifdef HAVE_ASYNC_LDS
    asm volatile("s_wait_asynccnt 0x0" ::: "memory");
#endif
    __syncthreads();
    const int i0 = threadIdx.x & 31;          // channel on write-out
    const int i1 = threadIdx.x >> 5;
#pragma unroll
    for (int r = 0; r < 4; ++r) {
        const int t = i1 + r * 8;
        XT[((unsigned)b * T + t0 + t) * Cpad + c0 + i0] = tobf(tilebuf[i0][t]);
    }
}

// Elementwise fp32 -> bf16 weight pack.
__global__ __launch_bounds__(256)
void k_packw(const float* __restrict__ W, __bf16* __restrict__ Wb, int n)
{
    const int i = blockIdx.x * 256 + threadIdx.x;
    if (i < n) Wb[i] = tobf(W[i]);
}

// Conv weight reshape: per layer (M,K,3) -> (3,M,K), fp32 -> bf16.
__global__ __launch_bounds__(256)
void k_packw3(const float* __restrict__ W, __bf16* __restrict__ Wb, int L, int M, int K)
{
    const int idx = blockIdx.x * 256 + threadIdx.x;
    if (idx >= L * M * K) return;
    const int l = idx / (M * K);
    const int rem = idx - l * M * K;
    const int m = rem / K;
    const int c = rem - m * K;
    const float* Wl = W + (size_t)l * M * K * 3;
    __bf16* Wbl = Wb + (size_t)l * M * K * 3;
#pragma unroll
    for (int j = 0; j < 3; ++j)
        Wbl[((unsigned)j * M + m) * K + c] = tobf(Wl[((unsigned)m * K + c) * 3 + j]);
}

// Pad weights (M,Kin) -> (M,Kpad) bf16, zero-filled.
__global__ __launch_bounds__(256)
void k_packwpad(const float* __restrict__ W, __bf16* __restrict__ Wb, int M, int Kin, int Kpad)
{
    const int idx = blockIdx.x * 256 + threadIdx.x;
    if (idx >= M * Kpad) return;
    const int m = idx / Kpad;
    const int k = idx - m * Kpad;
    Wb[idx] = tobf((k < Kin) ? W[(unsigned)m * Kin + k] : 0.0f);
}

// Per-channel batchnorm statistics over (B,T).
__global__ __launch_bounds__(256)
void k_bnstats(const float* __restrict__ X, float* __restrict__ mu, float* __restrict__ rs,
               int C, int T, int Bn, float eps)
{
    const int c = blockIdx.x;
    float s = 0.0f, s2 = 0.0f;
    for (int b = 0; b < Bn; ++b)
        for (int t = threadIdx.x; t < T; t += 256) {
            const float x = X[((unsigned)b * C + c) * T + t];
            s += x; s2 += x * x;
        }
    __shared__ float sh[512];
    sh[threadIdx.x] = s; sh[256 + threadIdx.x] = s2;
    __syncthreads();
    for (int o = 128; o > 0; o >>= 1) {
        if ((int)threadIdx.x < o) {
            sh[threadIdx.x] += sh[threadIdx.x + o];
            sh[256 + threadIdx.x] += sh[256 + threadIdx.x + o];
        }
        __syncthreads();
    }
    if (threadIdx.x == 0) {
        const int N = Bn * T;
        const float m = sh[0] / (float)N;
        const float v = sh[256] / (float)N - m * m;
        mu[c] = m;
        rs[c] = rsqrtf(v + eps);
    }
}

// Fold BN affine into Wq/bq; emit bf16 Weff directly.
__global__ __launch_bounds__(256)
void k_fold(const float* __restrict__ Wq, const float* __restrict__ bq,
            const float* __restrict__ g, const float* __restrict__ bb,
            const float* __restrict__ mu, const float* __restrict__ rs,
            __bf16* __restrict__ Weff, float* __restrict__ beff, int C)
{
    const int o = blockIdx.x;
    float acc = 0.0f;
    for (int c = threadIdx.x; c < C; c += 256) {
        const float s = g[c] * rs[c];
        const float w = Wq[(unsigned)o * C + c];
        Weff[(unsigned)o * C + c] = tobf(w * s);
        acc += w * (bb[c] - mu[c] * s);
    }
    __shared__ float sh[256];
    sh[threadIdx.x] = acc;
    __syncthreads();
    for (int off = 128; off > 0; off >>= 1) {
        if ((int)threadIdx.x < off) sh[threadIdx.x] += sh[threadIdx.x + off];
        __syncthreads();
    }
    if (threadIdx.x == 0) beff[o] = bq[o] + acc;
}

// Flash-style sliding-window cross attention + exact GELU (all-bf16 operands).
__global__ __launch_bounds__(256)
void k_att(const __bf16* __restrict__ qT, const __bf16* __restrict__ kT,
           const __bf16* __restrict__ Vb, __bf16* __restrict__ OT,
           int T, int d, float scale)
{
    const int lane = threadIdx.x & 31;
    const int wid  = threadIdx.x >> 5;
    const int half = lane >> 4;
    const int col  = lane & 15;
    const int tile = blockIdx.x * 8 + wid;
    const int tilesPerB = T / 16;
    const int b  = tile / tilesPerB;
    const int q0 = (tile % tilesPerB) * 16;
    const int CR = 128;
    const int p  = d >> 1;
    const int NC = (d >= 16) ? (d >> 4) : 1;
    const int win_start = (d >= 16) ? ((q0 / d) * d - p) : (q0 - p);
    const int nblk  = (q0 + col) / d;
    const int wbase = nblk * d - p;
    const int wlim  = 2 * d - 1;
    const unsigned bT = (unsigned)b * T;

    v16bf qf[4];
    {
        const __bf16* qr = qT + (unsigned)(bT + q0 + col) * CR + 8 * half;
#pragma unroll
        for (int ks = 0; ks < 4; ++ks) {
            const v8bf lo = *(const v8bf*)(qr + ks * 32);
            const v8bf hi = *(const v8bf*)(qr + ks * 32 + 16);
            qf[ks] = SHUF16(lo, hi);
        }
    }

    v8f oa[8];
#pragma unroll
    for (int i = 0; i < 8; ++i) { v8f z = {}; oa[i] = z; }
    float mrow = -3.0e38f, lrow = 0.0f;

    for (int c = 0; c < NC; ++c) {
        const int s0 = win_start + c * 32;
        const bool inb = (s0 >= 0) && (s0 + 32 <= T);
        float e0[8], e1[8];
#pragma unroll
        for (int tt = 0; tt < 2; ++tt) {
            v8f acc = {};
            const int tk = s0 + tt * 16 + col;
            const bool okK = (tk >= 0) && (tk < T);
            const int tkc = okK ? tk : 0;
            const __bf16* kr = kT + (unsigned)(bT + tkc) * CR + 8 * half;
            // Batch all 4 k-step loads ahead of the WMMA chain.
            v8bf klo[4], khi[4];
#pragma unroll
            for (int ks = 0; ks < 4; ++ks) {
                klo[ks] = *(const v8bf*)(kr + ks * 32);
                khi[ks] = *(const v8bf*)(kr + ks * 32 + 16);
            }
#pragma unroll
            for (int ks = 0; ks < 4; ++ks) {
                v16bf af = SHUF16(klo[ks], khi[ks]);
                const v16bf zf = {};
                af = okK ? af : zf;
                acc = __builtin_amdgcn_wmma_f32_16x16x32_bf16(false, af, false, qf[ks], (short)0, acc, false, false);
            }
            float* eo = tt ? e1 : e0;
#pragma unroll
            for (int v = 0; v < 8; ++v) {
                const int ca = c * 32 + tt * 16 + v + 8 * half;
                const int w  = (win_start + ca) - wbase;
                eo[v] = (w >= 0 && w < wlim) ? acc[v] * scale : -1.0e30f;
            }
        }
        float mx = -3.0e38f;
#pragma unroll
        for (int v = 0; v < 8; ++v) { mx = fmaxf(mx, e0[v]); mx = fmaxf(mx, e1[v]); }
        mx = fmaxf(mx, __shfl_xor(mx, 16, 32));
        const float mnew = fmaxf(mrow, mx);
        const float rescale = __expf(mrow - mnew);
        float a0[8], a1[8], ls = 0.0f;
#pragma unroll
        for (int v = 0; v < 8; ++v) {
            a0[v] = __expf(e0[v] - mnew);
            a1[v] = __expf(e1[v] - mnew);
            ls += a0[v] + a1[v];
        }
        ls += __shfl_xor(ls, 16, 32);
        lrow = lrow * rescale + ls;
        mrow = mnew;
        v16bf bfr;
#pragma unroll
        for (int e = 0; e < 8; ++e) { bfr[e] = tobf(a0[e]); bfr[e + 8] = tobf(a1[e]); }
        // Rescale all accumulators once, then V*att in two groups of 4 M-tiles
        // (loads batched ahead of the WMMAs).
#pragma unroll
        for (int mt = 0; mt < 8; ++mt)
#pragma unroll
            for (int v = 0; v < 8; ++v) oa[mt][v] *= rescale;
        const bool fastV = inb && (d >= 16);
#pragma unroll
        for (int g = 0; g < 2; ++g) {
            v16bf vf[4];
            if (fastV) {
                v8bf vlo[4], vhi[4];
#pragma unroll
                for (int i = 0; i < 4; ++i) {
                    const int row = (g * 4 + i) * 16 + col;
                    const __bf16* vr = Vb + (unsigned)(b * CR + row) * T + s0 + 8 * half;
                    vlo[i] = *(const v8bf*)(vr);
                    vhi[i] = *(const v8bf*)(vr + 16);
                }
#pragma unroll
                for (int i = 0; i < 4; ++i) vf[i] = SHUF16(vlo[i], vhi[i]);
            } else {
#pragma unroll
                for (int i = 0; i < 4; ++i) {
                    const int row = (g * 4 + i) * 16 + col;
                    const __bf16* vr = Vb + (unsigned)(b * CR + row) * T;
#pragma unroll
                    for (int e = 0; e < 16; ++e) {
                        const int t = s0 + KMAP(e, half);
                        const int tc = (t >= 0 && t < T) ? t : 0;
                        const __bf16 raw = vr[tc];
                        vf[i][e] = (t >= 0 && t < T) ? raw : (__bf16)0.0f;
                    }
                }
            }
#pragma unroll
            for (int i = 0; i < 4; ++i)
                oa[g * 4 + i] = __builtin_amdgcn_wmma_f32_16x16x32_bf16(false, vf[i], false, bfr, (short)0, oa[g * 4 + i], false, false);
        }
    }

    const float inv = (lrow > 0.0f) ? (1.0f / lrow) : 0.0f;
    __bf16* orow = OT + (unsigned)(bT + q0 + col) * CR + 8 * half;
#pragma unroll
    for (int mt = 0; mt < 8; ++mt) {
        v8bf pk;
#pragma unroll
        for (int v = 0; v < 8; ++v) pk[v] = tobf(gelu_f(oa[mt][v] * inv));
        *(v8bf*)(orow + mt * 16) = pk;
    }
}

extern "C" void kernel_launch(void* const* d_in, const int* in_sizes, int n_in,
                              void* d_out, int out_size, void* d_ws, size_t ws_size,
                              hipStream_t stream)
{
    (void)in_sizes; (void)n_in; (void)out_size; (void)ws_size;
    const int B = 4, T = 8192, C = 256, CR = 128, L = 10, IN = 48, INP = 64, NCLS = 48;

    const float* x     = (const float*)d_in[0];
    const float* fenc  = (const float*)d_in[1];
    const float* mask  = (const float*)d_in[2];
    const float* in_W  = (const float*)d_in[3];
    const float* in_b  = (const float*)d_in[4];
    const float* ff_W  = (const float*)d_in[5];
    const float* ff_b  = (const float*)d_in[6];
    const float* bn_g  = (const float*)d_in[7];
    const float* bn_b  = (const float*)d_in[8];
    const float* Wq    = (const float*)d_in[9];
    const float* bq    = (const float*)d_in[10];
    const float* Wk    = (const float*)d_in[11];
    const float* bk    = (const float*)d_in[12];
    const float* Wv    = (const float*)d_in[13];
    const float* bv    = (const float*)d_in[14];
    const float* Wo    = (const float*)d_in[15];
    const float* bo    = (const float*)d_in[16];
    const float* W1    = (const float*)d_in[17];
    const float* b1    = (const float*)d_in[18];
    const float* out_W = (const float*)d_in[19];
    const float* out_b = (const float*)d_in[20];

    const size_t BCT = (size_t)B * C * T;
    const size_t BRT = (size_t)B * CR * T;
    char* cur = (char*)d_ws;
    auto alignup = [&]() { cur = (char*)(((uintptr_t)cur + 255) & ~(uintptr_t)255); };
    auto allocF  = [&](size_t n) { alignup(); float*  p = (float*)cur;  cur += n * 4; return p; };
    auto allocBF = [&](size_t n) { alignup(); __bf16* p = (__bf16*)cur; cur += n * 2; return p; };
    float*  feat  = allocF(BCT);
    float*  conv  = allocF(BCT);
    __bf16* featT = allocBF(BCT);
    __bf16* convT = allocBF(BCT);
    __bf16* attpT = allocBF(BCT);
    __bf16* fencT = allocBF(BCT);
    __bf16* xT    = allocBF((size_t)B * T * INP);
    __bf16* qT    = allocBF(BRT);    // q, then attention output (in place)
    __bf16* kT    = allocBF(BRT);
    __bf16* vbf   = allocBF(BRT);
    __bf16* ffp   = allocBF((size_t)L * 3 * C * C);
    __bf16* wkp   = allocBF((size_t)L * CR * C);
    __bf16* wvp   = allocBF((size_t)L * CR * C);
    __bf16* wop   = allocBF((size_t)L * C * CR);
    __bf16* w1p   = allocBF((size_t)L * C * C);
    __bf16* inp   = allocBF((size_t)C * INP);
    __bf16* outp  = allocBF((size_t)NCLS * C);
    __bf16* weffb = allocBF((size_t)CR * C);
    float*  beff  = allocF(CR);
    float*  mu    = allocF(C);
    float*  rs    = allocF(C);

    const dim3 blk(256);
    auto gemmGrid = [&](int M) { return dim3(T / 512, M / 16, B); };
    auto packGrid = [&](int n) { return dim3((n + 255) / 256); };
    const float scale = 0.08838834764831845f;   // 1/sqrt(128)

    // ---- one-time packs ----
    k_pack<<<dim3(T / 32, C / 32, B), blk, 0, stream>>>(fenc, fencT, C, C, T);
    k_pack<<<dim3(T / 32, INP / 32, B), blk, 0, stream>>>(x, xT, IN, INP, T);
    k_packw3<<<packGrid(L * C * C), blk, 0, stream>>>(ff_W, ffp, L, C, C);
    k_packw<<<packGrid(L * CR * C), blk, 0, stream>>>(Wk, wkp, L * CR * C);
    k_packw<<<packGrid(L * CR * C), blk, 0, stream>>>(Wv, wvp, L * CR * C);
    k_packw<<<packGrid(L * C * CR), blk, 0, stream>>>(Wo, wop, L * C * CR);
    k_packw<<<packGrid(L * C * C), blk, 0, stream>>>(W1, w1p, L * C * C);
    k_packw<<<packGrid(NCLS * C), blk, 0, stream>>>(out_W, outp, NCLS * C);
    k_packwpad<<<packGrid(C * INP), blk, 0, stream>>>(in_W, inp, C, IN, INP);

    // feature = pw(x, in_W, in_b): zero-padded K=64
    k_gemm<<<gemmGrid(C), blk, 0, stream>>>(inp, in_b, xT, feat, featT, nullptr,
                                            nullptr, mask, C, INP, T, 1, F_WY | F_WT, 1.0f);

    for (int i = 0; i < L; ++i) {
        const int d = 1 << i;
        k_gemm<<<gemmGrid(C), blk, 0, stream>>>(ffp + (size_t)i * 3 * C * C, ff_b + i * C,
                                                featT, conv, convT, nullptr,
                                                nullptr, mask, C, C, T, d,
                                                F_GELU | F_CONV | F_WY | F_WT, 1.0f);
        k_bnstats<<<dim3(C), blk, 0, stream>>>(conv, mu, rs, C, T, B, 1e-5f);
        k_fold<<<dim3(CR), blk, 0, stream>>>(Wq + (size_t)i * CR * C, bq + i * CR,
                                             bn_g + i * C, bn_b + i * C, mu, rs, weffb, beff, C);
        k_gemm<<<gemmGrid(CR), blk, 0, stream>>>(weffb, beff, convT, nullptr, qT, nullptr,
                                                 nullptr, mask, CR, C, T, 1, F_WT, 1.0f);
        k_gemm<<<gemmGrid(CR), blk, 0, stream>>>(wkp + (size_t)i * CR * C, bk + i * CR,
                                                 fencT, nullptr, kT, nullptr,
                                                 nullptr, mask, CR, C, T, 1, F_WT, 1.0f);
        k_gemm<<<gemmGrid(CR), blk, 0, stream>>>(wvp + (size_t)i * CR * C, bv + i * CR,
                                                 fencT, nullptr, nullptr, vbf,
                                                 nullptr, mask, CR, C, T, 1, F_WB16, 1.0f);
        k_att<<<dim3(B * T / 16 / 8), blk, 0, stream>>>(qT, kT, vbf, qT, T, d, scale);
        k_gemm<<<gemmGrid(C), blk, 0, stream>>>(wop + (size_t)i * C * CR, bo + i * C,
                                                qT, nullptr, attpT, nullptr,
                                                conv, mask, C, CR, T, 1,
                                                F_MASKPRE | F_RES | F_WT, 1.0f);
        k_gemm<<<gemmGrid(C), blk, 0, stream>>>(w1p + (size_t)i * C * C, b1 + i * C,
                                                attpT, feat, featT, nullptr,
                                                feat, mask, C, C, T, 1,
                                                F_RES | F_MASKPOST | F_WY | F_WT, 1.0f);
    }

    float* logits = (float*)d_out;
    k_gemm<<<gemmGrid(NCLS), blk, 0, stream>>>(outp, out_b, featT, logits, nullptr, nullptr,
                                               nullptr, mask, NCLS, C, T, 1,
                                               F_MASKPOST | F_WY, 1.0f);
    (void)hipMemcpyAsync(logits + (size_t)B * NCLS * T, feat, BCT * sizeof(float),
                         hipMemcpyDeviceToDevice, stream);
}
